// MeanPredictor_44152263803376
// MI455X (gfx1250) — matrix operand order repaired and checked by reading, not compiled, not run on hardware
//
#include <hip/hip_runtime.h>
#include <hip/hip_bf16.h>

typedef __attribute__((ext_vector_type(16))) __bf16 v16bf;
typedef __attribute__((ext_vector_type(8)))  float  v8f;
typedef unsigned short u16;
typedef unsigned int   u32;

#define NN     12288
#define F_IN   256
#define F_HID  512
#define F_OUT  128
#define SLOPE  0.2f

// ---------- bf16 helpers (round-to-nearest-even) ----------
__device__ __forceinline__ u16 f2bf(float x) {
  u32 u = __float_as_uint(x);
  u32 r = u + 0x7FFFu + ((u >> 16) & 1u);
  return (u16)(r >> 16);
}
__device__ __forceinline__ float bf2f(u16 v) {
  return __uint_as_float(((u32)v) << 16);
}

union Frag { v16bf v; u32 u[8]; };

// A fragment 16x32 (MxK) bf16; LDS laid out [m][k] packed as dwords (16 dwords/row)
// ISA layout: lane m=lane&15, half=lane>>4; VGPR v: k = (v>>2)*16 + half*8 + (v&3)*2
__device__ __forceinline__ v16bf load_fragA(const u32* lds, int mBase, int lane) {
  int m = mBase + (lane & 15);
  int half = lane >> 4;
  Frag f;
#pragma unroll
  for (int v = 0; v < 8; ++v) {
    int k = ((v >> 2) * 16) + half * 8 + ((v & 3) * 2);
    f.u[v] = lds[m * 16 + (k >> 1)];
  }
  return f.v;
}

// B fragment 32x16 (KxN) bf16; LDS laid out [n][k] packed as dwords (16 dwords/row)
// ISA layout: lane n=lane&15, half=lane>>4; VGPR v: k = (v>>2)*16 + half*8 + (v&3)*2
__device__ __forceinline__ v16bf load_fragB(const u32* lds, int nBase, int lane) {
  int n = nBase + (lane & 15);
  int half = lane >> 4;
  Frag f;
#pragma unroll
  for (int v = 0; v < 8; ++v) {
    int k = ((v >> 2) * 16) + half * 8 + ((v & 3) * 2);
    f.u[v] = lds[n * 16 + (k >> 1)];
  }
  return f.v;
}

// ---------- f32 -> bf16 conversion with output stride ----------
__global__ void k_cvt(const float* __restrict__ src, u16* __restrict__ dst,
                      int rows, int cols, int ldd) {
  int idx = blockIdx.x * 256 + threadIdx.x;
  if (idx >= rows * cols) return;
  int r = idx / cols, c = idx - r * cols;
  dst[r * ldd + c] = f2bf(src[idx]);
}

// ---------- generic C = relu(A @ W^T + bias), bf16 in, f32 acc ----------
// A: [M x K] bf16 row-major (stride lda). W: [Nc x K] bf16 row-major.
// 64x64 block per workgroup, 256 threads = 8 waves, BK=32.
// Tile staging uses one B128 global load + one B128 LDS store per thread.
__global__ void k_gemm(const u16* __restrict__ A, int lda,
                       const u16* __restrict__ W, int K,
                       const float* __restrict__ bias,
                       u16* __restrict__ outB, int ldob,
                       float* __restrict__ outF, int ldof, int doRelu) {
  __shared__ alignas(16) u32 ldsA[64 * 16];
  __shared__ alignas(16) u32 ldsB[64 * 16];
  const int t = threadIdx.x;
  const int lane = t & 31, w = t >> 5;
  const int m0 = blockIdx.x * 64, n0 = blockIdx.y * 64;
  const int rt = w >> 1;            // row tile 0..3
  const int ct0 = (w & 1) * 32;     // two col tiles at ct0, ct0+16
  const int mrow = t >> 2;          // staging: row 0..63
  const int qk = (t & 3) * 8;       // staging: element offset within 32-wide K

  v8f acc0 = {}; v8f acc1 = {};

  for (int k0 = 0; k0 < K; k0 += 32) {
    __syncthreads();
    // A tile 64x32 -> one uint4 per thread
    uint4 va = *(const uint4*)(A + (m0 + mrow) * lda + k0 + qk);
    uint4 vb = *(const uint4*)(W + (n0 + mrow) * K + k0 + qk);
    *(uint4*)(&ldsA[4 * t]) = va;
    *(uint4*)(&ldsB[4 * t]) = vb;
    __syncthreads();

    v16bf af = load_fragA(ldsA, rt * 16, lane);
    v16bf b0 = load_fragB(ldsB, ct0, lane);
    v16bf b1 = load_fragB(ldsB, ct0 + 16, lane);
    acc0 = __builtin_amdgcn_wmma_f32_16x16x32_bf16(false, af, false, b0, (short)0, acc0, false, false);
    acc1 = __builtin_amdgcn_wmma_f32_16x16x32_bf16(false, af, false, b1, (short)0, acc1, false, false);
  }

  const int n = lane & 15, half = lane >> 4;
#pragma unroll
  for (int v = 0; v < 8; ++v) {
    int row = m0 + rt * 16 + v + half * 8;
#pragma unroll
    for (int s = 0; s < 2; ++s) {
      int col = n0 + ct0 + s * 16 + n;
      float x = (s == 0) ? acc0[v] : acc1[v];
      if (bias) x += bias[col];
      if (doRelu) x = fmaxf(x, 0.f);
      if (outB) outB[row * ldob + col] = f2bf(x);
      if (outF) outF[row * ldof + col] = x;
    }
  }
}

// ---------- f1 = Hh @ aL, f2 = Hh @ aR (one wave per row) ----------
__global__ void k_rowdot(const u16* __restrict__ Hh, const float* __restrict__ aL,
                         const float* __restrict__ aR,
                         float* __restrict__ f1, float* __restrict__ f2) {
  int lane = threadIdx.x & 31, w = threadIdx.x >> 5;
  int row = blockIdx.x * 8 + w;
  float s1 = 0.f, s2 = 0.f;
#pragma unroll
  for (int p = 0; p < 8; ++p) {
    int k = lane + p * 32;
    float h = bf2f(Hh[row * F_IN + k]);
    s1 += h * aL[k];
    s2 += h * aR[k];
  }
#pragma unroll
  for (int m = 16; m >= 1; m >>= 1) {
    s1 += __shfl_xor(s1, m, 32);
    s2 += __shfl_xor(s2, m, 32);
  }
  if (lane == 0) { f1[row] = s1; f2[row] = s2; }
}

// ---------- Fused flash-softmax Z = softmax_row(masked e) @ H ----------
// Single pass over A_norm (the 576MB array): online max/sum with WMMA
// accumulator rescaling between K tiles. 32 rows x 256 cols per workgroup.
// Result written as bf16 into Inph columns [256,512).
__global__ void k_zgemm(const float* __restrict__ Aadj, const u16* __restrict__ Hh,
                        const float* __restrict__ f1, const float* __restrict__ f2,
                        u16* __restrict__ Inph) {
  __shared__ alignas(16) u16 ldsAl[32 * 32];     // alpha tile [m][k] bf16
  __shared__ alignas(16) u32 ldsH[256 * 16];     // H tile [c][k-pairs]
  __shared__ float ldsScale[32];
  __shared__ float ldsL[32];
  const int t = threadIdx.x;
  const int lane = t & 31, w = t >> 5;
  const int half = lane >> 4;
  const int i0 = blockIdx.x * 32;
  const int tx = lane, my = w;          // staging: rows p*8+my (p<4), k-col = tx

  // per-thread online-softmax state for rows i0 + p*8 + my
  float pf1[4], mrun[4], lrun[4];
#pragma unroll
  for (int p = 0; p < 4; ++p) {
    pf1[p]  = f1[i0 + p * 8 + my];
    mrun[p] = -INFINITY;
    lrun[p] = 0.f;
  }

  const int rt = w & 1;                 // row tile 0..1
  const int cb = (w >> 1) * 64;         // 4 col tiles starting here
  v8f acc[4];
#pragma unroll
  for (int s = 0; s < 4; ++s) acc[s] = (v8f){};

  for (int j0 = 0; j0 < NN; j0 += 32) {
    __syncthreads();
    // ---- build alpha tile + update online softmax stats ----
    float f2j = f2[j0 + tx];
#pragma unroll
    for (int p = 0; p < 4; ++p) {
      int m = p * 8 + my;
      __builtin_prefetch(Aadj + (size_t)(i0 + m) * NN + j0 + 32 + tx, 0, 1);
      float a = Aadj[(size_t)(i0 + m) * NN + j0 + tx];
      float e = -INFINITY;
      if (a > 0.f) {
        e = pf1[p] + f2j;
        e = (e > 0.f) ? e : SLOPE * e;
      }
      float tmax = e;
#pragma unroll
      for (int o = 16; o >= 1; o >>= 1) tmax = fmaxf(tmax, __shfl_xor(tmax, o, 32));
      float mnew  = fmaxf(mrun[p], tmax);
      float scale = (mnew == -INFINITY) ? 1.f : __expf(mrun[p] - mnew);
      float al = (a > 0.f) ? __expf(e - mnew) : 0.f;
      float ps = al;
#pragma unroll
      for (int o = 16; o >= 1; o >>= 1) ps += __shfl_xor(ps, o, 32);
      lrun[p] = lrun[p] * scale + ps;
      mrun[p] = mnew;
      ldsAl[m * 32 + tx] = f2bf(al);
      if (lane == 0) ldsScale[m] = scale;
    }
    // ---- stage H tile [32 x 256] -> LDS [c][k-pairs], coalesced b32 loads ----
    {
      u16* lh = (u16*)ldsH;
      const int c2 = t & 127;           // column pair
      const int jh = t >> 7;            // 0..1
#pragma unroll
      for (int pp = 0; pp < 16; ++pp) {
        int jr = pp * 2 + jh;
        u32 u = *(const u32*)(Hh + (j0 + jr) * F_IN + 2 * c2);
        lh[(2 * c2) * 32 + jr]     = (u16)(u & 0xFFFFu);
        lh[(2 * c2 + 1) * 32 + jr] = (u16)(u >> 16);
      }
    }
    __syncthreads();

    // ---- rescale accumulators for this wave's rows, then WMMA ----
    float srow[8];
#pragma unroll
    for (int v = 0; v < 8; ++v) srow[v] = ldsScale[rt * 16 + v + 8 * half];
#pragma unroll
    for (int s = 0; s < 4; ++s)
#pragma unroll
      for (int v = 0; v < 8; ++v) acc[s][v] *= srow[v];

    v16bf af = load_fragA((const u32*)ldsAl, rt * 16, lane);
#pragma unroll
    for (int s = 0; s < 4; ++s) {
      v16bf bf = load_fragB(ldsH, cb + s * 16, lane);
      acc[s] = __builtin_amdgcn_wmma_f32_16x16x32_bf16(false, af, false, bf, (short)0, acc[s], false, false);
    }
  }

  // ---- publish running row sums, normalize, store bf16 into Inph ----
  if (lane == 0) {
#pragma unroll
    for (int p = 0; p < 4; ++p) ldsL[p * 8 + my] = lrun[p];
  }
  __syncthreads();
  const int n = lane & 15;
#pragma unroll
  for (int v = 0; v < 8; ++v) {
    int lr  = rt * 16 + v + 8 * half;
    float rinv = 1.f / ldsL[lr];
    int row = i0 + lr;
#pragma unroll
    for (int s = 0; s < 4; ++s) {
      int col = cb + s * 16 + n;
      Inph[row * (2 * F_IN) + F_IN + col] = f2bf(acc[s][v] * rinv);
    }
  }
}

// ====================================================================
extern "C" void kernel_launch(void* const* d_in, const int* in_sizes, int n_in,
                              void* d_out, int out_size, void* d_ws, size_t ws_size,
                              hipStream_t stream) {
  const float* X    = (const float*)d_in[0];
  const float* Aadj = (const float*)d_in[1];
  const float* Wgat = (const float*)d_in[2];
  const float* aL   = (const float*)d_in[3];
  const float* aR   = (const float*)d_in[4];
  const float* W1   = (const float*)d_in[5];
  const float* b1   = (const float*)d_in[6];
  const float* W2   = (const float*)d_in[7];
  const float* b2   = (const float*)d_in[8];
  float* out = (float*)d_out;

  char* ws = (char*)d_ws;
  size_t off = 0;
  auto alloc = [&](size_t bytes) -> void* {
    void* p = ws + off;
    off += (bytes + 255) & ~(size_t)255;
    return p;
  };
  u16* Inph  = (u16*)alloc((size_t)NN * 2 * F_IN * 2);   // [N,512] bf16: [X | Z]
  u16* Hh    = (u16*)alloc((size_t)NN * F_IN * 2);       // H_trans bf16
  u16* H1h   = (u16*)alloc((size_t)NN * F_HID * 2);      // relu(fc1) bf16
  u16* Wgath = (u16*)alloc((size_t)F_IN * F_IN * 2);
  u16* W1h   = (u16*)alloc((size_t)F_HID * 2 * F_IN * 2);
  u16* W2h   = (u16*)alloc((size_t)F_OUT * F_HID * 2);
  float* f1  = (float*)alloc((size_t)NN * 4);
  float* f2  = (float*)alloc((size_t)NN * 4);

  // 1) convert inputs to bf16 (X goes into Inph cols [0,256) with stride 512)
  k_cvt<<<(NN * F_IN + 255) / 256, 256, 0, stream>>>(X, Inph, NN, F_IN, 2 * F_IN);
  k_cvt<<<(F_IN * F_IN + 255) / 256, 256, 0, stream>>>(Wgat, Wgath, F_IN, F_IN, F_IN);
  k_cvt<<<(F_HID * 2 * F_IN + 255) / 256, 256, 0, stream>>>(W1, W1h, F_HID, 2 * F_IN, 2 * F_IN);
  k_cvt<<<(F_OUT * F_HID + 255) / 256, 256, 0, stream>>>(W2, W2h, F_OUT, F_HID, F_HID);

  // 2) H = X @ Wgat^T  (WMMA)
  k_gemm<<<dim3(NN / 64, F_IN / 64), 256, 0, stream>>>(
      Inph, 2 * F_IN, Wgath, F_IN, nullptr, Hh, F_IN, nullptr, 0, 0);

  // 3) f1 = H @ aL, f2 = H @ aR
  k_rowdot<<<NN / 8, 256, 0, stream>>>(Hh, aL, aR, f1, f2);

  // 4) fused flash-softmax + Z = alpha @ H (single 576MB pass over A_norm, WMMA)
  k_zgemm<<<NN / 32, 256, 0, stream>>>(Aadj, Hh, f1, f2, Inph);

  // 5) H1 = relu([X|Z] @ W1^T + b1)  (WMMA)
  k_gemm<<<dim3(NN / 64, F_HID / 64), 256, 0, stream>>>(
      Inph, 2 * F_IN, W1h, 2 * F_IN, b1, H1h, F_HID, nullptr, 0, 1);

  // 6) out = H1 @ W2^T + b2  (WMMA, f32 output)
  k_gemm<<<dim3(NN / 64, F_OUT / 64), 256, 0, stream>>>(
      H1h, F_HID, W2h, F_HID, b2, nullptr, 0, out, F_OUT, 0);

  (void)in_sizes; (void)n_in; (void)out_size; (void)ws_size;
}
